// TemplatePair_1279900254324
// MI455X (gfx1250) — matrix operand order repaired and checked by reading, not compiled, not run on hardware
//
#include <hip/hip_runtime.h>

// ---------------------------------------------------------------------------
// CDNA5 (gfx1250) WMMA-based implementation of the template-pair stack.
// All GEMMs run through v_wmma_f32_16x16x32_f16 (f16 in, f32 accumulate).
// wave32; 256-thread workgroups = 8 waves.  All WMMA fragments are fed by
// aligned 16-byte LDS loads (ds_load_b128) with K-contiguous B operands,
// and B/A fragments are software double-buffered to hide LDS latency.
// ---------------------------------------------------------------------------

typedef _Float16 h16;
typedef __attribute__((ext_vector_type(8)))  _Float16 v8h;
typedef __attribute__((ext_vector_type(16))) _Float16 v16h;
typedef __attribute__((ext_vector_type(8)))  float    v8f;

#define DEV __device__ __forceinline__

// A (16x32 f16), row-major source A[m][k] with pitch lda (multiple of 8):
// lane 0-15 -> M=lane, K={kb..kb+7, 16+kb..16+kb+7}, kb=0; lanes 16-31 kb=8.
// Two contiguous 8-half runs -> two ds_load_b128.
DEV v16h frag_a(const h16* A, int lda, int m0, int k0, int lane) {
  const int m  = m0 + (lane & 15);
  const int kb = (lane >> 4) << 3;                 // 0 or 8
  const h16* p = A + m * lda + k0 + kb;
  v8h lo = *(const v8h*)(p);
  v8h hi = *(const v8h*)(p + 16);
  return __builtin_shufflevector(lo, hi, 0, 1, 2, 3, 4, 5, 6, 7,
                                 8, 9, 10, 11, 12, 13, 14, 15);
}

// B (32x16 f16) from a K-CONTIGUOUS (n-major) source: src[n][k], pitch ld
// (multiple of 8).  Lane 0-15 -> col n, K={0..7,16..23}; lanes 16-31 -> K={8..15,24..31}.
// Two 16-byte runs -> two ds_load_b128.
DEV v16h frag_bt(const h16* Bt, int ld, int k0, int n0, int lane) {
  const int n  = n0 + (lane & 15);
  const int kh = (lane >> 4) << 3;                 // 0 or 8
  const h16* p = Bt + n * ld + k0 + kh;
  v8h lo = *(const v8h*)(p);
  v8h hi = *(const v8h*)(p + 16);
  return __builtin_shufflevector(lo, hi, 0, 1, 2, 3, 4, 5, 6, 7,
                                 8, 9, 10, 11, 12, 13, 14, 15);
}

DEV v8f wmma16(v16h a, v16h b, v8f c) {
  return __builtin_amdgcn_wmma_f32_16x16x32_f16(false, a, false, b,
                                                (short)0, c, false, false);
}

// ---------------------------------------------------------------------------
// LayerNorm: one wave32 per row; each lane owns 4 contiguous channels so the
// global loads vectorize.  f32 or f16 input, f16 output.  grid = M/8.
// ---------------------------------------------------------------------------
__global__ void ln_k(const float* __restrict__ xf, const h16* __restrict__ xh,
                     const float* __restrict__ gam, const float* __restrict__ bet,
                     h16* __restrict__ out, int C) {
  const int lane = threadIdx.x & 31;
  const int row  = blockIdx.x * 8 + (threadIdx.x >> 5);
  const int i0   = lane * 4;
  float vals[4];
  float s = 0.f;
#pragma unroll
  for (int t = 0; t < 4; ++t) {
    int i = i0 + t;
    float v = 0.f;
    if (i < C) v = xf ? xf[(size_t)row * C + i] : (float)xh[(size_t)row * C + i];
    vals[t] = v; s += v;
  }
#pragma unroll
  for (int off = 16; off; off >>= 1) s += __shfl_xor(s, off, 32);
  float mu = s / (float)C;
  float vs = 0.f;
#pragma unroll
  for (int t = 0; t < 4; ++t) {
    if (i0 + t < C) { float d = vals[t] - mu; vs += d * d; }
  }
#pragma unroll
  for (int off = 16; off; off >>= 1) vs += __shfl_xor(vs, off, 32);
  float rinv = rsqrtf(vs / (float)C + 1e-5f);
#pragma unroll
  for (int t = 0; t < 4; ++t) {
    int i = i0 + t;
    if (i < C)
      out[(size_t)row * C + i] = (h16)((vals[t] - mu) * rinv * gam[i] + bet[i]);
  }
}

// ---------------------------------------------------------------------------
// Generic fused GEMM:  Y[M,N] = epi( Xh[M,K](f16) @ W[K,N](f32) + bias )
// epi: act(0 none,1 sigmoid,2 relu), optional *mul(f16), +res(f32);
// output f32 (Yf) or f16 (Yh).  128x128 tile/WG.  K must be a multiple of 8
// (here: 88/128/256); padded to 32 in LDS.  W staged transposed (n-major).
// B (and next-K A) fragments are double-buffered; wrap-around indices keep
// the pipelined loads branch-free.
// ---------------------------------------------------------------------------
__global__ void gemm_k(const h16* __restrict__ X, const float* __restrict__ W,
                       const float* __restrict__ bias,
                       const h16* __restrict__ mul, const float* __restrict__ res,
                       float* __restrict__ Yf, h16* __restrict__ Yh,
                       int K, int Kpad, int N, int act) {
  extern __shared__ char smem[];
  const int ldx = Kpad + 8;                 // padded pitches (banks + alignment)
  h16* Xs = (h16*)smem;                     // 128 x ldx   (m-major)
  h16* Ws = Xs + 128 * ldx;                 // 128 x ldx   (n-major!)
  const int tid = threadIdx.x;
  const int m0  = blockIdx.x * 128;
  const int n0  = blockIdx.y * 128;
  const int kc  = Kpad >> 3;                // 8-half chunks per row

  // X tile: vector loads (16B), zero-pad chunks past K.
  for (int e = tid; e < 128 * kc; e += 256) {
    int r = e & 127, kk = (e >> 7) << 3;
    v8h vv = (v8h)(h16)0.f;
    if (kk < K) vv = *(const v8h*)(X + (size_t)(m0 + r) * K + kk);
    *(v8h*)(Xs + r * ldx + kk) = vv;
  }
  // W tile: transpose f32 -> f16 into n-major rows; global reads stay
  // coalesced across lanes (consecutive n), LDS stores are 16B vectors.
  for (int e = tid; e < 128 * kc; e += 256) {
    int n = e & 127, kk = (e >> 7) << 3;
    v8h vv;
#pragma unroll
    for (int u = 0; u < 8; ++u) {
      int k2 = kk + u;
      vv[u] = (k2 < K && (n0 + n) < N) ? (h16)W[(size_t)k2 * N + n0 + n] : (h16)0.f;
    }
    *(v8h*)(Ws + n * ldx + kk) = vv;
  }
  __syncthreads();

  const int wave = tid >> 5, lane = tid & 31;
  const int mw = wave * 16;
  v8f acc[8] = {};
  v16h a = frag_a(Xs, ldx, mw, 0, lane);
  v16h b = frag_bt(Ws, ldx, 0, 0, lane);
  for (int k0 = 0; k0 < Kpad; k0 += 32) {
    const int k0n = (k0 + 32 < Kpad) ? k0 + 32 : 0;   // wrap (harmless reload)
    v16h an = frag_a(Xs, ldx, mw, k0n, lane);
#pragma unroll
    for (int nt = 0; nt < 8; ++nt) {
      v16h bn = (nt < 7) ? frag_bt(Ws, ldx, k0, (nt + 1) * 16, lane)
                         : frag_bt(Ws, ldx, k0n, 0, lane);
      acc[nt] = wmma16(a, b, acc[nt]);
      b = bn;
    }
    a = an;
  }
  const int mb = (lane >> 4) << 3, nl = lane & 15;
#pragma unroll
  for (int nt = 0; nt < 8; ++nt) {
    int n = n0 + nt * 16 + nl;
    if (n >= N) continue;
#pragma unroll
    for (int r = 0; r < 8; ++r) {
      int m = m0 + mw + mb + r;
      float v2 = acc[nt][r];
      if (bias) v2 += bias[n];
      if (act == 1) v2 = 1.f / (1.f + __expf(-v2));
      else if (act == 2) v2 = v2 > 0.f ? v2 : 0.f;
      size_t idx = (size_t)m * N + n;
      if (mul) v2 *= (float)mul[idx];
      if (res) v2 += res[idx];
      if (Yh) Yh[idx] = (h16)v2; else Yf[idx] = v2;
    }
  }
}

// ---------------------------------------------------------------------------
// Triangle attention: one WG per (batch-image, line, head).
//   starting: rows of S are j (within row i=line); ending: rows are i.
// S = Q K^T * scale + bias -> softmax -> O = A V; gate applied at store.
// V staged d-major so O's B-fragments are K-contiguous.
// ---------------------------------------------------------------------------
#define LDQ 40     // 32 + 8 pad
#define LDV 136    // 128 + 8 pad
__global__ void attn_k(const h16* __restrict__ q, const h16* __restrict__ k,
                       const h16* __restrict__ v, const h16* __restrict__ g,
                       const float* __restrict__ bias, h16* __restrict__ o,
                       int starting) {
  extern __shared__ char smem[];
  h16*   Qs = (h16*)smem;                    // 128 x LDQ  (row x d)
  h16*   Ks = Qs + 128 * LDQ;                // 128 x LDQ  (row x d)
  h16*   Vt = Ks + 128 * LDQ;                // 32  x LDV  (d   x row)
  float* Sf = (float*)(Vt + 32 * LDV);       // 128 x 128 f32
  h16*   Ah = (h16*)(Sf + 128 * 128);        // 128 x LDV f16

  const int tid  = threadIdx.x;
  const int h    = blockIdx.x & 3;
  const int line = (blockIdx.x >> 2) & 127;
  const int bi   = blockIdx.x >> 9;
  const int base = bi * 16384;

  // Stage Q/K (16B vector loads) and V transposed.
  for (int e = tid; e < 512; e += 256) {     // 512 = 128 rows * 4 chunks
    int t = e >> 2, d8 = (e & 3) << 3;
    int row = base + (starting ? line * 128 + t : t * 128 + line);
    size_t idx = (size_t)row * 128 + h * 32 + d8;
    *(v8h*)(Qs + t * LDQ + d8) = *(const v8h*)(q + idx);
    *(v8h*)(Ks + t * LDQ + d8) = *(const v8h*)(k + idx);
    v8h vv = *(const v8h*)(v + idx);
#pragma unroll
    for (int u = 0; u < 8; ++u) Vt[(d8 + u) * LDV + t] = vv[u];
  }
  __syncthreads();

  const int wave = tid >> 5, lane = tid & 31;
  const int j0 = wave * 16;
  const int mb = (lane >> 4) << 3, nl = lane & 15;
  {
    v16h a = frag_a(Qs, LDQ, j0, 0, lane);   // D=32 == one WMMA K step
    v16h b = frag_bt(Ks, LDQ, 0, 0, lane);
    const float scale = 0.1767766953f;       // 1/sqrt(32)
#pragma unroll
    for (int kt = 0; kt < 8; ++kt) {
      v16h bn = frag_bt(Ks, LDQ, 0, ((kt + 1) & 7) * 16, lane);  // pipelined
      v8f acc = {};
      acc = wmma16(a, b, acc);
      b = bn;
#pragma unroll
      for (int r = 0; r < 8; ++r)
        Sf[(j0 + mb + r) * 128 + kt * 16 + nl] = acc[r] * scale;
    }
  }
  __syncthreads();

  if (tid < 128) {                            // f32 softmax over k (+pair bias)
    int m = tid;
    size_t bb = ((size_t)(base + m * 128)) * 4 + h;
    float mx = -3.0e38f;
    for (int kk = 0; kk < 128; ++kk) {
      float s2 = Sf[m * 128 + kk] + bias[bb + (size_t)kk * 4];
      Sf[m * 128 + kk] = s2;
      mx = fmaxf(mx, s2);
    }
    float sum = 0.f;
    for (int kk = 0; kk < 128; ++kk) {
      float e2 = __expf(Sf[m * 128 + kk] - mx);
      Sf[m * 128 + kk] = e2;
      sum += e2;
    }
    float rs = 1.f / sum;
    for (int kk = 0; kk < 128; ++kk)
      Ah[m * LDV + kk] = (h16)(Sf[m * 128 + kk] * rs);
  }
  __syncthreads();

  v8f acc2[2] = {};
  {
    v16h a = frag_a(Ah, LDV, j0, 0, lane);
    v16h b = frag_bt(Vt, LDV, 0, 0, lane);
#pragma unroll
    for (int ks = 0; ks < 4; ++ks) {
      const int k0 = ks * 32;
      const int k0n = ((ks + 1) & 3) * 32;
      v16h an = frag_a(Ah, LDV, j0, k0n, lane);
#pragma unroll
      for (int nt = 0; nt < 2; ++nt) {
        v16h bn = (nt == 0) ? frag_bt(Vt, LDV, k0, 16, lane)
                            : frag_bt(Vt, LDV, k0n, 0, lane);
        acc2[nt] = wmma16(a, b, acc2[nt]);
        b = bn;
      }
      a = an;
    }
  }
#pragma unroll
  for (int nt = 0; nt < 2; ++nt)
#pragma unroll
    for (int r = 0; r < 8; ++r) {
      int m = j0 + mb + r;
      int row = base + (starting ? line * 128 + m : m * 128 + line);
      size_t idx = (size_t)row * 128 + h * 32 + nt * 16 + nl;
      o[idx] = (h16)(acc2[nt][r] * (float)g[idx]);
    }
}

// ---------------------------------------------------------------------------
// Triangle multiplication einsum: one WG per (batch-image, channel c).
// outgoing: vals[i,j] = sum_k a[i,k,c]*b[j,k,c]; incoming uses a[k,i],b[k,j].
// ---------------------------------------------------------------------------
#define LDT 136    // 128 + 8 pad
__global__ void trimul_k(const h16* __restrict__ a, const h16* __restrict__ b,
                         h16* __restrict__ vals, int outgoing) {
  extern __shared__ char smem[];
  h16* As = (h16*)smem;          // 128 x LDT (i x k)
  h16* Bs = As + 128 * LDT;      // 128 x LDT (j x k)
  const int tid = threadIdx.x;
  const int c   = blockIdx.x & 127;
  const int bi  = blockIdx.x >> 7;
  const size_t base = (size_t)bi * 2097152;   // 128*128*128

  for (int e = tid; e < 128 * 128; e += 256) {
    int i = e >> 7, k2 = e & 127;
    size_t ia = outgoing ? base + (((size_t)i * 128 + k2) << 7) + c
                         : base + (((size_t)k2 * 128 + i) << 7) + c;
    As[i * LDT + k2] = a[ia];
    Bs[i * LDT + k2] = b[ia];    // identical index pattern with j := i
  }
  __syncthreads();

  const int wave = tid >> 5, lane = tid & 31;
  const int i0 = wave * 16;
  v8f acc[8] = {};
  v16h af = frag_a(As, LDT, i0, 0, lane);
  v16h bf = frag_bt(Bs, LDT, 0, 0, lane);
#pragma unroll
  for (int ks = 0; ks < 4; ++ks) {
    const int k0 = ks * 32;
    const int k0n = ((ks + 1) & 3) * 32;
    v16h an = frag_a(As, LDT, i0, k0n, lane);
#pragma unroll
    for (int jt = 0; jt < 8; ++jt) {
      v16h bn = (jt < 7) ? frag_bt(Bs, LDT, k0, (jt + 1) * 16, lane)
                         : frag_bt(Bs, LDT, k0n, 0, lane);
      acc[jt] = wmma16(af, bf, acc[jt]);
      bf = bn;
    }
    af = an;
  }
  const int mb = (lane >> 4) << 3, nl = lane & 15;
#pragma unroll
  for (int jt = 0; jt < 8; ++jt)
#pragma unroll
    for (int r = 0; r < 8; ++r) {
      size_t idx = base + (((size_t)(i0 + mb + r) * 128 + jt * 16 + nl) << 7) + c;
      vals[idx] = (h16)acc[jt][r];
    }
}

// ---------------------------------------------------------------------------
// Host orchestration
// ---------------------------------------------------------------------------
static const int M_ROWS = 65536;   // B*T*N*N = 4*128*128

static void launch_gemm(hipStream_t st, const h16* X, int K,
                        const float* W, const float* bias, int N, int act,
                        const h16* mul, const float* res, float* Yf, h16* Yh) {
  int Kpad = (K + 31) & ~31;
  dim3 grid(M_ROWS / 128, (N + 127) / 128);
  size_t sh = (size_t)512 * (Kpad + 8);    // 2 * 128 * (Kpad+8) halfs
  gemm_k<<<grid, 256, sh, st>>>(X, W, bias, mul, res, Yf, Yh, K, Kpad, N, act);
}

extern "C" void kernel_launch(void* const* d_in, const int* in_sizes, int n_in,
                              void* d_out, int out_size, void* d_ws, size_t ws_size,
                              hipStream_t stream) {
  (void)in_sizes; (void)n_in; (void)out_size; (void)ws_size;

  // Allow >64KB dynamic LDS (WGP has 320KB)
  hipFuncSetAttribute((const void*)gemm_k,   hipFuncAttributeMaxDynamicSharedMemorySize, 160 * 1024);
  hipFuncSetAttribute((const void*)attn_k,   hipFuncAttributeMaxDynamicSharedMemorySize, 160 * 1024);
  hipFuncSetAttribute((const void*)trimul_k, hipFuncAttributeMaxDynamicSharedMemorySize, 160 * 1024);

  auto F = [&](int i) { return (const float*)d_in[i]; };
  int ix = 0;
  const float* tfeat    = F(ix++);          // (1,4,128,128,88)
  const float* ln_in_g  = F(ix++);
  const float* ln_in_b  = F(ix++);
  const float* lin_in_w = F(ix++);
  const float* lin_in_b = F(ix++);

  struct Att { const float *ln_g,*ln_b,*wq,*wk,*wv,*wb,*gw,*gb,*ow,*ob; };
  struct Mlt { const float *lnp_g,*lnp_b,*lno_g,*lno_b,*g1w,*g1b,*g2w,*g2b,
                           *l1w,*l1b,*l2w,*l2b,*gw,*gb,*ow,*ob; };
  struct Trn { const float *ln_g,*ln_b,*uw,*ub,*dw,*db; };
  Att as_[2], ae_[2]; Mlt mo_[2], mi_[2]; Trn tr_[2];
  for (int bl = 0; bl < 2; ++bl) {
    auto rdA = [&](Att& A){ A.ln_g=F(ix++);A.ln_b=F(ix++);A.wq=F(ix++);A.wk=F(ix++);
                            A.wv=F(ix++);A.wb=F(ix++);A.gw=F(ix++);A.gb=F(ix++);
                            A.ow=F(ix++);A.ob=F(ix++); };
    auto rdM = [&](Mlt& P){ P.lnp_g=F(ix++);P.lnp_b=F(ix++);P.lno_g=F(ix++);P.lno_b=F(ix++);
                            P.g1w=F(ix++);P.g1b=F(ix++);P.g2w=F(ix++);P.g2b=F(ix++);
                            P.l1w=F(ix++);P.l1b=F(ix++);P.l2w=F(ix++);P.l2b=F(ix++);
                            P.gw=F(ix++);P.gb=F(ix++);P.ow=F(ix++);P.ob=F(ix++); };
    rdA(as_[bl]); rdA(ae_[bl]); rdM(mo_[bl]); rdM(mi_[bl]);
    Trn& T = tr_[bl];
    T.ln_g=F(ix++);T.ln_b=F(ix++);T.uw=F(ix++);T.ub=F(ix++);T.dw=F(ix++);T.db=F(ix++);
  }

  // Workspace layout (all per M_ROWS rows):
  const size_t MB = 1024 * 1024;
  char* w   = (char*)d_ws;
  h16*  zh  = (h16*)(w);             // 16MB: LN output (Mx128 or Mx88)
  h16*  s1  = (h16*)(w + 16 * MB);   // 32MB: q / a / transition hidden (Mx256)
  h16*  s2  = (h16*)(w + 48 * MB);   // 16MB: k / b
  h16*  s3  = (h16*)(w + 64 * MB);   // 16MB: v / mult gate
  h16*  s4  = (h16*)(w + 80 * MB);   // 16MB: attn gate / LN(vals)
  h16*  s5  = (h16*)(w + 96 * MB);   // 16MB: o / vals
  float* bb = (float*)(w + 112 * MB);// 1MB : attention pair bias (Mx4)
  float* x  = (float*)d_out;         // 32MB: running pair state

  auto ln = [&](const float* xf, const h16* xh, const float* gg,
                const float* be, h16* out, int C) {
    ln_k<<<dim3(M_ROWS / 8), 256, 0, stream>>>(xf, xh, gg, be, out, C);
  };

  const size_t attn_lds = (size_t)(2 * 128 * LDQ + 32 * LDV) * 2
                        + 128 * 128 * 4 + (size_t)128 * LDV * 2;  // 129536 B
  auto attn = [&](const Att& A, int starting) {
    ln(x, nullptr, A.ln_g, A.ln_b, zh, 128);
    launch_gemm(stream, zh, 128, A.wq, nullptr, 128, 0, nullptr, nullptr, nullptr, s1);
    launch_gemm(stream, zh, 128, A.wk, nullptr, 128, 0, nullptr, nullptr, nullptr, s2);
    launch_gemm(stream, zh, 128, A.wv, nullptr, 128, 0, nullptr, nullptr, nullptr, s3);
    launch_gemm(stream, zh, 128, A.gw, A.gb,   128, 1, nullptr, nullptr, nullptr, s4);
    launch_gemm(stream, zh, 128, A.wb, nullptr,  4, 0, nullptr, nullptr, bb, nullptr);
    attn_k<<<dim3(4 * 128 * 4), 256, attn_lds, stream>>>(s1, s2, s3, s4, bb, s5, starting);
    launch_gemm(stream, s5, 128, A.ow, A.ob,   128, 0, nullptr, x, x, nullptr);
  };

  auto mult = [&](const Mlt& P, int outgoing) {
    ln(x, nullptr, P.lnp_g, P.lnp_b, zh, 128);
    launch_gemm(stream, zh, 128, P.l1w, P.l1b, 128, 0, nullptr, nullptr, nullptr, s1);
    launch_gemm(stream, zh, 128, P.g1w, P.g1b, 128, 1, s1,      nullptr, nullptr, s1); // a
    launch_gemm(stream, zh, 128, P.l2w, P.l2b, 128, 0, nullptr, nullptr, nullptr, s2);
    launch_gemm(stream, zh, 128, P.g2w, P.g2b, 128, 1, s2,      nullptr, nullptr, s2); // b
    launch_gemm(stream, zh, 128, P.gw,  P.gb,  128, 1, nullptr, nullptr, nullptr, s3); // g
    trimul_k<<<dim3(512), 256, (size_t)2 * 128 * LDT * 2, stream>>>(s1, s2, s5, outgoing);
    ln(nullptr, s5, P.lno_g, P.lno_b, s4, 128);
    launch_gemm(stream, s4, 128, P.ow, P.ob,   128, 0, s3, x, x, nullptr);
  };

  auto trans = [&](const Trn& T) {
    ln(x, nullptr, T.ln_g, T.ln_b, zh, 128);
    launch_gemm(stream, zh, 128, T.uw, T.ub, 256, 2, nullptr, nullptr, nullptr, s1);
    launch_gemm(stream, s1, 256, T.dw, T.db, 128, 0, nullptr, x, x, nullptr);
  };

  // input embedding: x = LN(template_feat) @ W + b
  ln(tfeat, nullptr, ln_in_g, ln_in_b, zh, 88);
  launch_gemm(stream, zh, 88, lin_in_w, lin_in_b, 128, 0, nullptr, nullptr, x, nullptr);

  for (int bl = 0; bl < 2; ++bl) {
    attn(as_[bl], 1);
    attn(ae_[bl], 0);
    mult(mo_[bl], 1);
    mult(mi_[bl], 0);
    trans(tr_[bl]);
  }
}
